// Mamba2_6614249636549
// MI455X (gfx1250) — compile-verified, matrix operand order checked by reading, not run on hardware
//
#include <hip/hip_runtime.h>
#include <hip/hip_bf16.h>

// ---------------------------------------------------------------------------
// Mamba2 SSD block for MI455X (gfx1250, wave32, WMMA f32_16x16x32_f16)
// All WMMA operands staged in LDS with K contiguous -> 2x ds_load_b128 / frag.
// ---------------------------------------------------------------------------

typedef __attribute__((ext_vector_type(16))) _Float16 v16h;
typedef __attribute__((ext_vector_type(8)))  _Float16 v8h;
typedef __attribute__((ext_vector_type(8)))  float    v8f;

static constexpr int BATCH = 2;
static constexpr int SEQ   = 2048;
static constexpr int H     = 32;
static constexpr int P     = 64;
static constexpr int HID   = 2048;
static constexpr int CHUNK = 256;
static constexpr int NC    = SEQ / CHUNK;   // 8 chunks
static constexpr int ROWS  = BATCH * SEQ;   // 4096 flat (b,s) rows

// ---------------- WMMA helpers -------------------------------------------
__device__ __forceinline__ v8f wmma_f16(v16h a, v16h b, v8f c) {
  return __builtin_amdgcn_wmma_f32_16x16x32_f16(false, a, false, b, (short)0, c,
                                                false, false);
}

// Fragment from an [M][K] array (A operand) or [N][K] array (B operand).
// lane l: major = base + (l&15); K-halves {kb..kb+7, kb+16..kb+23}, kb=8*(l>=16).
// ld must be a multiple of 8 halves and the array 16B-aligned -> 2x b128 loads.
__device__ __forceinline__ v16h ldfrag(const _Float16* p, int ld, int mbase,
                                       int kbase) {
  const int lane = threadIdx.x & 31;
  const _Float16* r = p + (size_t)(mbase + (lane & 15)) * ld + kbase +
                      ((lane >> 4) << 3);
  v8h lo = *(const v8h*)r;
  v8h hi = *(const v8h*)(r + 16);
  return __builtin_shufflevector(lo, hi, 0, 1, 2, 3, 4, 5, 6, 7, 8, 9, 10, 11,
                                 12, 13, 14, 15);
}

// C/D tile: VGPR r -> rows (r, r+8 for upper lane half), lane&15 -> column
__device__ __forceinline__ void st_tile(float* p, int ld, int rbase, int cbase,
                                        v8f acc) {
  const int lane = threadIdx.x & 31;
  float* o = p + (size_t)(rbase + ((lane >> 4) << 3)) * ld + cbase + (lane & 15);
#pragma unroll
  for (int r = 0; r < 8; ++r) o[(size_t)r * ld] = acc[r];
}

__device__ __forceinline__ v8f ld_tile(const float* p, int ld, int rbase,
                                       int cbase) {
  const int lane = threadIdx.x & 31;
  const float* o = p + (size_t)(rbase + ((lane >> 4) << 3)) * ld + cbase +
                   (lane & 15);
  v8f a;
#pragma unroll
  for (int r = 0; r < 8; ++r) a[r] = o[(size_t)r * ld];
  return a;
}

__device__ __forceinline__ float softplus_f(float x) {
  return x > 20.f ? x : log1pf(expf(x));
}

// ---------------- K1: dt / gate projections (WMMA) -------------------------
__global__ void k1_proj(const float* __restrict__ hab,
                        const float* __restrict__ hg,
                        const float* __restrict__ dt_w,
                        const float* __restrict__ dt_b,
                        const float* __restrict__ g_w,
                        const float* __restrict__ g_b,
                        const float* __restrict__ dt_bias,
                        float* __restrict__ ws_dt,
                        float* __restrict__ ws_gate) {
  const int tid = threadIdx.x;
  const int rows0 = blockIdx.x * 64;
  extern __shared__ __align__(16) char smem[];
  _Float16* sAab = (_Float16*)smem;       // 64 x 40   [row][k]
  _Float16* sAg  = sAab + 64 * 40;        // 64 x 40
  _Float16* sWd  = sAg + 64 * 40;         // 32 x 40   [h][k] (transposed)
  _Float16* sWg  = sWd + 32 * 40;         // 32 x 40

  const int w = tid >> 5, lane = tid & 31;
  const int strip = w & 3;                // 16-row strip
  const int sel = w >> 2;                 // 0 = dt, 1 = gate
  v8f z = {};
  v8f acc0 = z, acc1 = z;

  // per-thread staging slot: 8 contiguous halves of the 64x32 hidden tile
  const int ar = tid >> 2, ac = (tid & 3) * 8;

  for (int k0 = 0; k0 < HID; k0 += 32) {
    __syncthreads();
    {
      const float* s0 = hab + (size_t)(rows0 + ar) * HID + k0 + ac;
      const float* s1 = hg  + (size_t)(rows0 + ar) * HID + k0 + ac;
      v8h t0, t1;
#pragma unroll
      for (int e = 0; e < 8; ++e) { t0[e] = (_Float16)s0[e]; t1[e] = (_Float16)s1[e]; }
      *(v8h*)(sAab + ar * 40 + ac) = t0;
      *(v8h*)(sAg  + ar * 40 + ac) = t1;
    }
    for (int i = tid; i < 32 * 32; i += 256) {
      int kk = i >> 5, hh = i & 31;       // transposed store: [h][k]
      sWd[hh * 40 + kk] = (_Float16)dt_w[(size_t)(k0 + kk) * H + hh];
      sWg[hh * 40 + kk] = (_Float16)g_w [(size_t)(k0 + kk) * H + hh];
    }
    __syncthreads();
    const _Float16* sA = sel ? sAg : sAab;
    const _Float16* sW = sel ? sWg : sWd;
    v16h a  = ldfrag(sA, 40, strip * 16, 0);
    v16h b0 = ldfrag(sW, 40, 0, 0);
    v16h b1 = ldfrag(sW, 40, 16, 0);
    acc0 = wmma_f16(a, b0, acc0);
    acc1 = wmma_f16(a, b1, acc1);
  }

  const int row = rows0 + strip * 16 + ((lane >> 4) << 3);
  const int col = lane & 15;
  if (sel == 0) {
#pragma unroll
    for (int r = 0; r < 8; ++r) {
      float v0 = acc0[r] + dt_b[col]      + dt_bias[col];
      float v1 = acc1[r] + dt_b[col + 16] + dt_bias[col + 16];
      ws_dt[(size_t)(row + r) * H + col]      = softplus_f(v0);
      ws_dt[(size_t)(row + r) * H + col + 16] = softplus_f(v1);
    }
  } else {
#pragma unroll
    for (int r = 0; r < 8; ++r) {
      ws_gate[(size_t)(row + r) * H + col]      = acc0[r] + g_b[col];
      ws_gate[(size_t)(row + r) * H + col + 16] = acc1[r] + g_b[col + 16];
    }
  }
}

// ---------------- K2: per-(b,chunk,head) states + intra-chunk y ------------
__global__ void k2_chunk(const float* __restrict__ qs,
                         const float* __restrict__ ks,
                         const float* __restrict__ vs,
                         const float* __restrict__ ws_dt,
                         const float* __restrict__ A_log,
                         float* __restrict__ ws_cum,
                         float* __restrict__ ws_cd,
                         float* __restrict__ ws_S,
                         float* __restrict__ ws_y) {
  const int tid = threadIdx.x;
  const int bid = blockIdx.x;                 // (b*NC + c)*H + h
  const int h = bid % H;
  const int c = (bid / H) % NC;
  const int b = bid / (H * NC);

  extern __shared__ __align__(16) char smem[];
  _Float16* sC   = (_Float16*)smem;           // 256 x 72  C  [l][n]   (A: CB)
  _Float16* sB   = sC + 256 * 72;             // 256 x 72  B  [s][n]   (B: CB)
  _Float16* sBt  = sB + 256 * 72;             // 64 x 264  B^T [n][l]  (A: states)
  _Float16* sVt  = sBt + 64 * 264;            // 64 x 264  V^T [p][s]  (B: y)
  _Float16* sWXt = sVt + 64 * 264;            // 64 x 264  (w*x)^T [p][l] (B: states)
  _Float16* sS   = sWXt;                      // overlay: 8 waves x 16 x 136 scores
  float* sYred = (float*)(sWXt + 17408);      // 64 x 68 f32
  float* sCum  = sYred + 64 * 68;             // 256
  float* sDt   = sCum + 256;                  // 256
  float* sScan = sDt + 256;                   // 256

  // ---- phase 0: load rows, l2norm q/k, dt cumsum -------------------------
  {
    const int row = tid;                      // one row per thread
    const size_t sglob = (size_t)(b * SEQ + c * CHUNK + row);
    const size_t gb = (sglob * H + h) * P;
    float buf[64];

    // C = l2norm(q), contiguous [l][n]
    {
      const float4* p4 = reinterpret_cast<const float4*>(qs + gb);
      float ss = 0.f;
#pragma unroll
      for (int i = 0; i < 16; ++i) {
        float4 t = p4[i];
        buf[4*i] = t.x; buf[4*i+1] = t.y; buf[4*i+2] = t.z; buf[4*i+3] = t.w;
        ss += t.x*t.x + t.y*t.y + t.z*t.z + t.w*t.w;
      }
      float inv = 1.f / fmaxf(sqrtf(ss), 1e-12f);
#pragma unroll
      for (int i = 0; i < 8; ++i) {
        v8h t;
#pragma unroll
        for (int e = 0; e < 8; ++e) t[e] = (_Float16)(buf[8*i+e] * inv);
        *(v8h*)(sC + row * 72 + 8 * i) = t;
      }
    }
    // B = l2norm(k): row-major [s][n] AND transposed [n][l]
    {
      const float4* p4 = reinterpret_cast<const float4*>(ks + gb);
      float ss = 0.f;
#pragma unroll
      for (int i = 0; i < 16; ++i) {
        float4 t = p4[i];
        buf[4*i] = t.x; buf[4*i+1] = t.y; buf[4*i+2] = t.z; buf[4*i+3] = t.w;
        ss += t.x*t.x + t.y*t.y + t.z*t.z + t.w*t.w;
      }
      float inv = 1.f / fmaxf(sqrtf(ss), 1e-12f);
#pragma unroll
      for (int i = 0; i < 8; ++i) {
        v8h t;
#pragma unroll
        for (int e = 0; e < 8; ++e) t[e] = (_Float16)(buf[8*i+e] * inv);
        *(v8h*)(sB + row * 72 + 8 * i) = t;
      }
#pragma unroll
      for (int i = 0; i < 64; ++i)
        sBt[i * 264 + row] = (_Float16)(buf[i] * inv);
    }
    // V transposed [p][s]; keep row in buf for wx below
    {
      const float4* p4 = reinterpret_cast<const float4*>(vs + gb);
#pragma unroll
      for (int i = 0; i < 16; ++i) {
        float4 t = p4[i];
        buf[4*i] = t.x; buf[4*i+1] = t.y; buf[4*i+2] = t.z; buf[4*i+3] = t.w;
      }
#pragma unroll
      for (int i = 0; i < 64; ++i) sVt[i * 264 + row] = (_Float16)buf[i];
    }

    // dt, a = dt*A, inclusive scan over 256 rows
    const float dtv = ws_dt[sglob * H + h];
    const float Ah = -expf(A_log[h]);
    sScan[row] = dtv * Ah;
    __syncthreads();
    for (int off = 1; off < 256; off <<= 1) {
      float x = sScan[row];
      float yp = (row >= off) ? sScan[row - off] : 0.f;
      __syncthreads();
      sScan[row] = x + yp;
      __syncthreads();
    }
    const float cum = sScan[row];
    sCum[row] = cum;
    sDt[row] = dtv;
    ws_cum[sglob * H + h] = cum;
    __syncthreads();
    const float cumlast = sCum[255];
    if (row == 255) ws_cd[bid] = expf(cumlast);
    const float wr = expf(cumlast - cum) * dtv;
#pragma unroll
    for (int i = 0; i < 64; ++i)
      sWXt[i * 264 + row] = (_Float16)(wr * buf[i]);   // (w*x)^T [p][l]
  }
  __syncthreads();

  const int w = tid >> 5, lane = tid & 31;

  // ---- phase A: chunk state  S[n][p] = sum_l Bt[n][l] * wx[l][p] ---------
  {
    const int nt = w >> 1;
    const int ptb = (w & 1) * 2;
    v8f z = {};
    v8f a0 = z, a1 = z;
#pragma unroll
    for (int kk = 0; kk < CHUNK; kk += 32) {
      v16h af = ldfrag(sBt, 264, nt * 16, kk);
      v16h b0 = ldfrag(sWXt, 264, ptb * 16, kk);
      v16h b1 = ldfrag(sWXt, 264, ptb * 16 + 16, kk);
      a0 = wmma_f16(af, b0, a0);
      a1 = wmma_f16(af, b1, a1);
    }
    float* Sg = ws_S + (size_t)bid * 4096;
    st_tile(Sg, 64, nt * 16, ptb * 16, a0);
    st_tile(Sg, 64, nt * 16, ptb * 16 + 16, a1);
  }
  __syncthreads();   // sWXt dead -> safe to overlay score strips

  // ---- phase B: intra-chunk y (flash style, 4 row-blocks of 64) ----------
  const int strip = w & 3;       // 16-row strip inside the 64-row block
  const int half = w >> 2;       // 128-column half of the chunk
  _Float16* sSw = sS + (size_t)w * (16 * 136);

  for (int rb = 0; rb < 4; ++rb) {
    const int row0 = rb * 64;
    v8f z = {};
    v8f sc[8];
#pragma unroll
    for (int i = 0; i < 8; ++i) sc[i] = z;

#pragma unroll
    for (int kk = 0; kk < 64; kk += 32) {
      v16h a = ldfrag(sC, 72, row0 + strip * 16, kk);
#pragma unroll
      for (int ct = 0; ct < 8; ++ct) {
        v16h bf = ldfrag(sB, 72, half * 128 + ct * 16, kk);
        sc[ct] = wmma_f16(a, bf, sc[ct]);
      }
    }

    // decay * causal mask * dt, convert to f16 score strip [l][s]
    const int roff = (lane >> 4) << 3;
    const int cl = lane & 15;
#pragma unroll
    for (int ct = 0; ct < 8; ++ct) {
      const int sg = half * 128 + ct * 16 + cl;
      const float dts = sDt[sg];
      const float cums = sCum[sg];
#pragma unroll
      for (int r = 0; r < 8; ++r) {
        const int lg = row0 + strip * 16 + roff + r;
        float val = (sg <= lg) ? sc[ct][r] * expf(sCum[lg] - cums) * dts : 0.f;
        sSw[(roff + r) * 136 + ct * 16 + cl] = (_Float16)val;
      }
    }

    // y_partial = scores(16x128) @ V(128x64)
    v8f ya[4];
#pragma unroll
    for (int i = 0; i < 4; ++i) ya[i] = z;
#pragma unroll
    for (int kk = 0; kk < 128; kk += 32) {
      v16h a = ldfrag(sSw, 136, 0, kk);
#pragma unroll
      for (int pt = 0; pt < 4; ++pt) {
        v16h bf = ldfrag(sVt, 264, pt * 16, half * 128 + kk);
        ya[pt] = wmma_f16(a, bf, ya[pt]);
      }
    }

    __syncthreads();
    if (half == 0) {
#pragma unroll
      for (int pt = 0; pt < 4; ++pt)
        st_tile(sYred, 68, strip * 16, pt * 16, ya[pt]);
    }
    __syncthreads();
    if (half == 1) {
      float* yg = ws_y +
          ((size_t)(b * SEQ + c * CHUNK + row0 + strip * 16) * H + h) * P;
#pragma unroll
      for (int pt = 0; pt < 4; ++pt) {
        v8f o = ld_tile(sYred, 68, strip * 16, pt * 16);
#pragma unroll
        for (int r = 0; r < 8; ++r) o[r] += ya[pt][r];
        st_tile(yg, H * P, 0, pt * 16, o);
      }
    }
    __syncthreads();
  }
}

// ---------------- K3: inter-chunk sequential scan --------------------------
__global__ void k3_scan(const float* __restrict__ ws_S,
                        const float* __restrict__ ws_cd,
                        float* __restrict__ ws_Hp) {
  const int tid = threadIdx.x;
  const int b = blockIdx.x >> 5;
  const int h = blockIdx.x & 31;
  float Hst[16];
#pragma unroll
  for (int i = 0; i < 16; ++i) Hst[i] = 0.f;
  for (int c = 0; c < NC; ++c) {
    const size_t base = ((size_t)(b * NC + c) * H + h) * 4096;
    const float cd = ws_cd[(b * NC + c) * H + h];
#pragma unroll
    for (int i = 0; i < 16; ++i) {
      const size_t idx = base + tid + 256 * i;
      float s = ws_S[idx];
      ws_Hp[idx] = Hst[i];            // state BEFORE this chunk
      Hst[i] = cd * Hst[i] + s;
    }
  }
}

// ---------------- K4a: inter-chunk contribution (WMMA) ---------------------
__global__ void k4a_inter(const float* __restrict__ qs,
                          const float* __restrict__ ws_cum,
                          const float* __restrict__ ws_Hp,
                          float* __restrict__ ws_y) {
  const int tid = threadIdx.x;
  const int bid = blockIdx.x;
  const int h = bid % H;
  const int c = (bid / H) % NC;
  const int b = bid / (H * NC);

  extern __shared__ __align__(16) char smem[];
  _Float16* sCe  = (_Float16*)smem;     // 256 x 72 : C * exp(cum)  [l][n]
  _Float16* sHpT = sCe + 256 * 72;      // 64 x 72  : H_prev^T [p][n]

  {
    const int row = tid;
    const size_t sglob = (size_t)(b * SEQ + c * CHUNK + row);
    const size_t gb = (sglob * H + h) * P;
    const float4* p4 = reinterpret_cast<const float4*>(qs + gb);
    float buf[64];
    float ss = 0.f;
#pragma unroll
    for (int i = 0; i < 16; ++i) {
      float4 t = p4[i];
      buf[4*i] = t.x; buf[4*i+1] = t.y; buf[4*i+2] = t.z; buf[4*i+3] = t.w;
      ss += t.x*t.x + t.y*t.y + t.z*t.z + t.w*t.w;
    }
    float sc = expf(ws_cum[sglob * H + h]) / fmaxf(sqrtf(ss), 1e-12f);
#pragma unroll
    for (int i = 0; i < 8; ++i) {
      v8h t;
#pragma unroll
      for (int e = 0; e < 8; ++e) t[e] = (_Float16)(buf[8*i+e] * sc);
      *(v8h*)(sCe + row * 72 + 8 * i) = t;
    }

    for (int i = tid; i < 4096; i += 256) {       // transpose Hp -> [p][n]
      int n = i >> 6, p = i & 63;
      sHpT[p * 72 + n] = (_Float16)ws_Hp[(size_t)bid * 4096 + i];
    }
  }
  __syncthreads();

  const int w = tid >> 5;
  v8f z = {};
  for (int rt = 2 * w; rt < 2 * w + 2; ++rt) {
    v8f acc[4];
#pragma unroll
    for (int i = 0; i < 4; ++i) acc[i] = z;
#pragma unroll
    for (int kk = 0; kk < 64; kk += 32) {
      v16h a = ldfrag(sCe, 72, rt * 16, kk);
#pragma unroll
      for (int pt = 0; pt < 4; ++pt) {
        v16h bf = ldfrag(sHpT, 72, pt * 16, kk);
        acc[pt] = wmma_f16(a, bf, acc[pt]);
      }
    }
    float* yg = ws_y + ((size_t)(b * SEQ + c * CHUNK + rt * 16) * H + h) * P;
#pragma unroll
    for (int pt = 0; pt < 4; ++pt) {
      v8f o = ld_tile(yg, H * P, 0, pt * 16);
#pragma unroll
      for (int r = 0; r < 8; ++r) o[r] += acc[pt][r];
      st_tile(yg, H * P, 0, pt * 16, o);
    }
  }
}

// ---------------- K4b: gate (silu) + RMSNorm -> f16 ------------------------
__global__ void k4b_norm(const float* __restrict__ ws_y,
                         const float* __restrict__ ws_gate,
                         const float* __restrict__ norm_w,
                         _Float16* __restrict__ ws_yn) {
  const int w = threadIdx.x >> 5, lane = threadIdx.x & 31;
  const int row = blockIdx.x * 8 + w;           // flat (b,s)
  const float* yr = ws_y + (size_t)row * (H * P);
  float vals[64];
  float ss = 0.f;
#pragma unroll
  for (int i = 0; i < 64; ++i) {
    const int j = i * 32 + lane;
    const float g = ws_gate[(size_t)row * H + (j >> 6)];
    const float sg = g / (1.f + expf(-g));
    const float yv = yr[j] * sg;
    vals[i] = yv;
    ss += yv * yv;
  }
#pragma unroll
  for (int off = 16; off > 0; off >>= 1) ss += __shfl_xor(ss, off, 32);
  const float scale = rsqrtf(ss * (1.f / (H * P)) + 1e-5f);
  _Float16* on = ws_yn + (size_t)row * (H * P);
#pragma unroll
  for (int i = 0; i < 64; ++i) {
    const int j = i * 32 + lane;
    on[j] = (_Float16)(vals[i] * scale * norm_w[j]);
  }
}

// ---------------- K4c: per-head o_proj (WMMA) + bias -----------------------
__global__ void k4c_oproj(const _Float16* __restrict__ ws_yn,
                          const float* __restrict__ o_w,
                          const float* __restrict__ o_b,
                          float* __restrict__ out) {
  const int tid = threadIdx.x;
  const int h = blockIdx.x & 31;
  const int rows0 = (blockIdx.x >> 5) * 64;

  extern __shared__ __align__(16) char smem[];
  _Float16* sY  = (_Float16*)smem;    // 64 x 72  y_norm [l][p]
  _Float16* sWt = sY + 64 * 72;       // 64 x 72  o_w[h]^T [q][p]

  // y rows: contiguous f16 in global -> v8h copies
#pragma unroll
  for (int g = 0; g < 2; ++g) {
    const int idx = (g * 256 + tid) * 8;
    const int r = idx >> 6, cc = idx & 63;
    *(v8h*)(sY + r * 72 + cc) =
        *(const v8h*)(ws_yn + (size_t)(rows0 + r) * (H * P) + h * P + cc);
  }
  for (int i = tid; i < 64 * 64; i += 256) {   // transpose o_w -> [q][p]
    int r = i >> 6, cc = i & 63;
    sWt[cc * 72 + r] = (_Float16)o_w[(size_t)h * 4096 + i];
  }
  __syncthreads();

  const int w = tid >> 5, lane = tid & 31;
  const int mt = w >> 1;
  const int ntb = (w & 1) * 2;
  v8f z = {};
  v8f a0 = z, a1 = z;
#pragma unroll
  for (int kk = 0; kk < 64; kk += 32) {
    v16h a  = ldfrag(sY, 72, mt * 16, kk);
    v16h b0 = ldfrag(sWt, 72, ntb * 16, kk);
    v16h b1 = ldfrag(sWt, 72, ntb * 16 + 16, kk);
    a0 = wmma_f16(a, b0, a0);
    a1 = wmma_f16(a, b1, a1);
  }

  const int roff = (lane >> 4) << 3;
  const int q0 = ntb * 16 + (lane & 15);
  const float bias0 = o_b[h * P + q0];
  const float bias1 = o_b[h * P + q0 + 16];
  float* og = out + (size_t)(rows0 + mt * 16) * (H * P) + h * P;
#pragma unroll
  for (int r = 0; r < 8; ++r) {
    og[(size_t)(roff + r) * (H * P) + q0]      = a0[r] + bias0;
    og[(size_t)(roff + r) * (H * P) + q0 + 16] = a1[r] + bias1;
  }
}

// ---------------------------------------------------------------------------
extern "C" void kernel_launch(void* const* d_in, const int* in_sizes, int n_in,
                              void* d_out, int out_size, void* d_ws,
                              size_t ws_size, hipStream_t stream) {
  (void)in_sizes; (void)n_in; (void)out_size; (void)ws_size;
  const float* hab     = (const float*)d_in[0];
  const float* hg      = (const float*)d_in[1];
  const float* qs      = (const float*)d_in[2];
  const float* ks      = (const float*)d_in[3];
  const float* vs      = (const float*)d_in[4];
  const float* dt_w    = (const float*)d_in[5];
  const float* dt_b    = (const float*)d_in[6];
  const float* g_w     = (const float*)d_in[7];
  const float* g_b     = (const float*)d_in[8];
  const float* A_log   = (const float*)d_in[9];
  const float* dt_bias = (const float*)d_in[10];
  const float* norm_w  = (const float*)d_in[11];
  const float* o_w     = (const float*)d_in[12];
  const float* o_b     = (const float*)d_in[13];
  float* out = (float*)d_out;

  // workspace partition
  size_t off = 0;
  auto carve = [&](size_t bytes) {
    void* p = (char*)d_ws + off;
    off += (bytes + 255) & ~(size_t)255;
    return p;
  };
  float*     ws_dt   = (float*)carve((size_t)ROWS * H * 4);
  float*     ws_gate = (float*)carve((size_t)ROWS * H * 4);
  float*     ws_cum  = (float*)carve((size_t)ROWS * H * 4);
  float*     ws_cd   = (float*)carve((size_t)BATCH * NC * H * 4);
  float*     ws_S    = (float*)carve((size_t)BATCH * NC * H * 4096 * 4);
  float*     ws_Hp   = (float*)carve((size_t)BATCH * NC * H * 4096 * 4);
  float*     ws_y    = (float*)carve((size_t)ROWS * H * P * 4);
  _Float16*  ws_yn   = (_Float16*)carve((size_t)ROWS * H * P * 2);

  // K1: projections
  {
    const size_t shmem = (size_t)(64 * 40 * 2 + 32 * 40 * 2) * sizeof(_Float16);
    k1_proj<<<ROWS / 64, 256, shmem, stream>>>(hab, hg, dt_w, dt_b, g_w, g_b,
                                               dt_bias, ws_dt, ws_gate);
  }
  // K2: chunk states + intra-chunk y
  {
    const size_t shmem =
        (size_t)(256 * 72 * 2 + 64 * 264 * 2 + 17408) * sizeof(_Float16) +
        (size_t)(64 * 68 + 3 * 256) * sizeof(float);
    k2_chunk<<<BATCH * NC * H, 256, shmem, stream>>>(qs, ks, vs, ws_dt, A_log,
                                                     ws_cum, ws_cd, ws_S, ws_y);
  }
  // K3: sequential chunk scan
  k3_scan<<<BATCH * H, 256, 0, stream>>>(ws_S, ws_cd, ws_Hp);
  // K4a: inter-chunk contribution
  {
    const size_t shmem = (size_t)(256 * 72 + 64 * 72) * sizeof(_Float16);
    k4a_inter<<<BATCH * NC * H, 256, shmem, stream>>>(qs, ws_cum, ws_Hp, ws_y);
  }
  // K4b: gating + RMSNorm
  k4b_norm<<<ROWS / 8, 256, 0, stream>>>(ws_y, ws_gate, norm_w, ws_yn);
  // K4c: per-head o_proj
  {
    const size_t shmem = (size_t)(2 * 64 * 72) * sizeof(_Float16);
    k4c_oproj<<<(ROWS / 64) * H, 256, shmem, stream>>>(ws_yn, o_w, o_b, out);
  }
}
